// FCNet_31464930410614
// MI455X (gfx1250) — matrix operand order repaired and checked by reading, not compiled, not run on hardware
//
#include <hip/hip_runtime.h>
#include <hip/hip_bf16.h>

typedef __attribute__((ext_vector_type(16))) __bf16 v16bf;
typedef __attribute__((ext_vector_type(8)))  __bf16 v8bf;
typedef __attribute__((ext_vector_type(8)))  float  v8f;

__device__ __forceinline__ __bf16 f2bf(float f) {
    union { float f; unsigned u; } x; x.f = f;
    unsigned r = x.u + 0x7fffu + ((x.u >> 16) & 1u);   // round-to-nearest-even
    unsigned short h = (unsigned short)(r >> 16);
    return __builtin_bit_cast(__bf16, h);
}

// ---------------------------------------------------------------------------
// Pack weights W[deg][cout][cin][kh][kw] (f32) -> Wpack[kh*kw][npad][deg*cin] (bf16)
// K index = p*cin + c  (matches reference transpose(1,0,2,3,4).reshape merge)
// ---------------------------------------------------------------------------
__global__ void pack_weights(const float* __restrict__ W, __bf16* __restrict__ out,
                             int deg, int cout, int cin, int kh, int kw, int npad) {
    const int CE = deg * cin;
    const int total = kh * kw * npad * CE;
    int id = blockIdx.x * blockDim.x + threadIdx.x;
    if (id >= total) return;
    int k = id % CE;
    int n = (id / CE) % npad;
    int t = id / (CE * npad);
    int p = k / cin, c = k % cin;
    int khi = t / kw, kwi = t % kw;
    float v = 0.0f;
    if (n < cout)
        v = W[((((size_t)p * cout + n) * cin + c) * kh + khi) * kw + kwi];
    out[id] = f2bf(v);
}

// ---------------------------------------------------------------------------
// Layer 1: degree-4 poly conv (Cin=1, 5x5) fused with 2x2 maxpool + relu,
// emitting bf16 power-expanded NHWC activations [B][126*126][64]  (p*16+co)
// ---------------------------------------------------------------------------
__global__ __launch_bounds__(256)
void layer1_pool_pow(const float* __restrict__ x, const float* __restrict__ W1,
                     const float* __restrict__ b1, __bf16* __restrict__ act1) {
    const int HP = 126, WP = 126, WI = 256;
    int id = blockIdx.x * blockDim.x + threadIdx.x;
    const int total = 64 * HP * WP;
    if (id >= total) return;
    int ow = id % WP;
    int oh = (id / WP) % HP;
    int b  = id / (WP * HP);

    float pr[36];
    const float* xb = x + ((size_t)b * 256 + 2 * oh) * WI + 2 * ow;
#pragma unroll
    for (int r = 0; r < 6; ++r)
#pragma unroll
        for (int c = 0; c < 6; ++c)
            pr[r * 6 + c] = xb[(size_t)r * WI + c];

    size_t obase = (size_t)id * 64;
    for (int co = 0; co < 16; ++co) {
        float a0 = b1[co], a1 = a0, a2 = a0, a3 = a0;
#pragma unroll
        for (int kh = 0; kh < 5; ++kh)
#pragma unroll
            for (int kw = 0; kw < 5; ++kw) {
                const float w1 = W1[(0 * 16 + co) * 25 + kh * 5 + kw];
                const float w2 = W1[(1 * 16 + co) * 25 + kh * 5 + kw];
                const float w3 = W1[(2 * 16 + co) * 25 + kh * 5 + kw];
                const float w4 = W1[(3 * 16 + co) * 25 + kh * 5 + kw];
                {
                    float xv = pr[kh * 6 + kw]; float x2 = xv * xv;
                    a0 = fmaf(w4, x2 * x2, fmaf(w3, x2 * xv, fmaf(w2, x2, fmaf(w1, xv, a0))));
                }
                {
                    float xv = pr[kh * 6 + kw + 1]; float x2 = xv * xv;
                    a1 = fmaf(w4, x2 * x2, fmaf(w3, x2 * xv, fmaf(w2, x2, fmaf(w1, xv, a1))));
                }
                {
                    float xv = pr[(kh + 1) * 6 + kw]; float x2 = xv * xv;
                    a2 = fmaf(w4, x2 * x2, fmaf(w3, x2 * xv, fmaf(w2, x2, fmaf(w1, xv, a2))));
                }
                {
                    float xv = pr[(kh + 1) * 6 + kw + 1]; float x2 = xv * xv;
                    a3 = fmaf(w4, x2 * x2, fmaf(w3, x2 * xv, fmaf(w2, x2, fmaf(w1, xv, a3))));
                }
            }
        float a  = fmaxf(fmaxf(fmaxf(a0, a1), fmaxf(a2, a3)), 0.0f);
        float a2v = a * a;
        act1[obase + 0  + co] = f2bf(a);
        act1[obase + 16 + co] = f2bf(a2v);
        act1[obase + 32 + co] = f2bf(a2v * a);
        act1[obase + 48 + co] = f2bf(a2v * a2v);
    }
}

// ---------------------------------------------------------------------------
// Tap-split implicit-GEMM conv on WMMA bf16.
//   act  : [B][Hi*Wi][CE]     bf16 (NHWC, CE = power-expanded channels)
//   wpk  : [KH*KW][NPAD][CE]  bf16
// One wave -> 32 pixels x NPAD couts (2 x NT accumulators): each A fragment
// feeds NT WMMAs, each B fragment feeds 2 WMMAs; 2*NT independent D->C chains.
// Epilogue: POW=true  -> bf16 powers [pix][4*NPAD]  (requires NPAD==cout)
//           POW=false -> f32 y       [pix][NPAD]
// ---------------------------------------------------------------------------
template <int KH, int KW, int CE, int NPAD, bool POW>
__global__ __launch_bounds__(256)
void konv_wmma(const __bf16* __restrict__ act, const __bf16* __restrict__ wpk,
               const float* __restrict__ bias, int cout,
               void* __restrict__ outp,
               int Hi, int Wi, int Ho, int Wo) {
    constexpr int NT = NPAD / 16;            // cout subtiles per wave
    const int Mtot  = Ho * Wo;
    const int wid   = threadIdx.x >> 5;
    const int lane  = threadIdx.x & 31;
    const int b     = blockIdx.y;
    const int tbase = (blockIdx.x * 8 + wid) * 32;   // 32 pixels per wave
    if (tbase >= Mtot) return;               // wave-uniform exit: EXEC stays full

    const int row   = lane & 15;
    const int khalf = lane >> 4;

    int m0 = tbase + row;       if (m0 > Mtot - 1) m0 = Mtot - 1;
    int m1 = tbase + 16 + row;  if (m1 > Mtot - 1) m1 = Mtot - 1;
    const int h0a = m0 / Wo, w0a = m0 % Wo;
    const int h0b = m1 / Wo, w0b = m1 % Wo;

    const v8f vzero = {0.f, 0.f, 0.f, 0.f, 0.f, 0.f, 0.f, 0.f};
    v8f acc[2][NT];
#pragma unroll
    for (int mt = 0; mt < 2; ++mt)
#pragma unroll
        for (int nt = 0; nt < NT; ++nt) acc[mt][nt] = vzero;

    union Frag { v16bf v; struct { v8bf lo, hi; } h; };

#pragma unroll
    for (int kh = 0; kh < KH; ++kh) {
#pragma unroll
        for (int kw = 0; kw < KW; ++kw) {
            const __bf16* apa = act + (((size_t)b * Hi + (h0a + kh)) * Wi + (w0a + kw)) * CE
                                    + khalf * 8;
            const __bf16* apb = act + (((size_t)b * Hi + (h0b + kh)) * Wi + (w0b + kw)) * CE
                                    + khalf * 8;
            const __bf16* bp  = wpk + ((size_t)(kh * KW + kw) * NPAD + row) * CE
                                    + khalf * 8;
#pragma unroll
            for (int kb = 0; kb < CE; kb += 32) {
                Frag A0, A1, Bf[NT];
                A0.h.lo = *(const v8bf*)(apa + kb);        // K = kb+khalf*8 .. +7
                A0.h.hi = *(const v8bf*)(apa + kb + 16);   // K = kb+16+khalf*8 .. +7
                A1.h.lo = *(const v8bf*)(apb + kb);
                A1.h.hi = *(const v8bf*)(apb + kb + 16);
#pragma unroll
                for (int nt = 0; nt < NT; ++nt) {
                    Bf[nt].h.lo = *(const v8bf*)(bp + (size_t)nt * 16 * CE + kb);
                    Bf[nt].h.hi = *(const v8bf*)(bp + (size_t)nt * 16 * CE + kb + 16);
                }
#pragma unroll
                for (int nt = 0; nt < NT; ++nt) {
                    acc[0][nt] = __builtin_amdgcn_wmma_f32_16x16x32_bf16(
                                    false, A0.v, false, Bf[nt].v, (short)0,
                                    acc[0][nt], false, false);
                    acc[1][nt] = __builtin_amdgcn_wmma_f32_16x16x32_bf16(
                                    false, A1.v, false, Bf[nt].v, (short)0,
                                    acc[1][nt], false, false);
                }
            }
        }
    }

#pragma unroll
    for (int nt = 0; nt < NT; ++nt) {
        const int n = nt * 16 + row;
        const float bv = (n < cout) ? bias[n] : 0.0f;
#pragma unroll
        for (int mt = 0; mt < 2; ++mt) {
#pragma unroll
            for (int r = 0; r < 8; ++r) {
                int pix = tbase + mt * 16 + r + khalf * 8;  // D: vgpr r, lanes>=16 -> M+8
                if (pix < Mtot) {
                    float v = acc[mt][nt][r] + bv;
                    if (POW) {
                        __bf16* o = (__bf16*)outp
                                  + ((size_t)b * Mtot + pix) * (4 * NPAD) + n;
                        float v2 = v * v;
                        o[0]        = f2bf(v);
                        o[NPAD]     = f2bf(v2);
                        o[2 * NPAD] = f2bf(v2 * v);
                        o[3 * NPAD] = f2bf(v2 * v2);
                    } else {
                        ((float*)outp)[((size_t)b * Mtot + pix) * NPAD + n] = v;
                    }
                }
            }
        }
    }
}

// ---------------------------------------------------------------------------
// Mean over spatial: y[B][M][NPAD] -> out[B][cout]
// ---------------------------------------------------------------------------
__global__ __launch_bounds__(256)
void reduce_mean(const float* __restrict__ y, float* __restrict__ out,
                 int Mtot, int npad, int cout) {
    const int b = blockIdx.x;
    const int n = blockIdx.y;
    __shared__ float sm[256];
    float s = 0.0f;
    for (int i = threadIdx.x; i < Mtot; i += 256)
        s += y[((size_t)b * Mtot + i) * npad + n];
    sm[threadIdx.x] = s;
    __syncthreads();
    for (int k = 128; k > 0; k >>= 1) {
        if (threadIdx.x < k) sm[threadIdx.x] += sm[threadIdx.x + k];
        __syncthreads();
    }
    if (threadIdx.x == 0) out[(size_t)b * cout + n] = sm[0] / (float)Mtot;
}

// ---------------------------------------------------------------------------
extern "C" void kernel_launch(void* const* d_in, const int* in_sizes, int n_in,
                              void* d_out, int out_size, void* d_ws, size_t ws_size,
                              hipStream_t stream) {
    const float* x  = (const float*)d_in[0];
    const float* W1 = (const float*)d_in[1];
    const float* b1 = (const float*)d_in[2];
    const float* W2 = (const float*)d_in[3];
    const float* b2 = (const float*)d_in[4];
    const float* W3 = (const float*)d_in[5];
    const float* b3 = (const float*)d_in[6];
    const float* W4 = (const float*)d_in[7];
    const float* b4 = (const float*)d_in[8];
    float* out = (float*)d_out;

    const int B = 64;
    const int M1 = 126 * 126;            // pooled pixels
    const int M2 = 122 * 122;            // after 5x5
    const int M3 = 120 * 120;            // after 3x3
    const int M4 = 118 * 118;            // after 3x3

    // ---- workspace layout (with reuse) ----
    char* ws = (char*)d_ws;
    size_t off = 0;
    auto alloc = [&](size_t bytes) { size_t o = off; off = (off + bytes + 255) & ~(size_t)255; return o; };

    // region A: act1 (130MB) then reused as pow3 (236MB) — lifetimes disjoint
    size_t szAct1 = (size_t)B * M1 * 64  * sizeof(__bf16);
    size_t szPow3 = (size_t)B * M3 * 128 * sizeof(__bf16);
    size_t offA   = alloc(szAct1 > szPow3 ? szAct1 : szPow3);
    size_t offP2  = alloc((size_t)B * M2 * 128 * sizeof(__bf16));   // pow2
    size_t offY4  = alloc((size_t)B * M4 * 16  * sizeof(float));    // layer-4 f32 out
    size_t offW2  = alloc((size_t)25 * 32 * 64  * sizeof(__bf16));
    size_t offW3  = alloc((size_t)9  * 32 * 128 * sizeof(__bf16));
    size_t offW4  = alloc((size_t)9  * 16 * 128 * sizeof(__bf16));
    (void)ws_size;

    __bf16* act1 = (__bf16*)(ws + offA);
    __bf16* pow3 = (__bf16*)(ws + offA);
    __bf16* pow2 = (__bf16*)(ws + offP2);
    float*  y4   = (float*)(ws + offY4);
    __bf16* wp2  = (__bf16*)(ws + offW2);
    __bf16* wp3  = (__bf16*)(ws + offW3);
    __bf16* wp4  = (__bf16*)(ws + offW4);

    // ---- pack weights (tiny) ----
    {
        int t2 = 25 * 32 * 64;
        pack_weights<<<(t2 + 255) / 256, 256, 0, stream>>>(W2, wp2, 4, 32, 16, 5, 5, 32);
        int t3 = 9 * 32 * 128;
        pack_weights<<<(t3 + 255) / 256, 256, 0, stream>>>(W3, wp3, 4, 32, 32, 3, 3, 32);
        int t4 = 9 * 16 * 128;
        pack_weights<<<(t4 + 255) / 256, 256, 0, stream>>>(W4, wp4, 4, 10, 32, 3, 3, 16);
    }

    // ---- layer 1: direct conv + pool + relu + bf16 powers ----
    {
        int total = B * M1;
        layer1_pool_pow<<<(total + 255) / 256, 256, 0, stream>>>(x, W1, b1, act1);
    }

    // ---- layer 2: 5x5, CE=64, NPAD=32, fused bf16-power epilogue ----
    {
        dim3 grid((M2 + 255) / 256, B);
        konv_wmma<5, 5, 64, 32, true><<<grid, 256, 0, stream>>>(
            act1, wp2, b2, 32, pow2, 126, 126, 122, 122);
    }

    // ---- layer 3: 3x3, CE=128, NPAD=32, fused bf16-power epilogue ----
    {
        dim3 grid((M3 + 255) / 256, B);
        konv_wmma<3, 3, 128, 32, true><<<grid, 256, 0, stream>>>(
            pow2, wp3, b3, 32, pow3, 122, 122, 120, 120);
    }

    // ---- layer 4: 3x3, CE=128, NPAD=16 (cout=10 zero-padded), f32 out ----
    {
        dim3 grid((M4 + 255) / 256, B);
        konv_wmma<3, 3, 128, 16, false><<<grid, 256, 0, stream>>>(
            pow3, wp4, b4, 10, y4, 120, 120, 118, 118);
    }

    // ---- mean over spatial -> [64,10] ----
    {
        dim3 grid(B, 10);
        reduce_mean<<<grid, 256, 0, stream>>>(y4, out, M4, 16, 10);
    }
}
